// HQLinear_30021821399184
// MI455X (gfx1250) — matrix operand order repaired and checked by reading, not compiled, not run on hardware
//
#include <hip/hip_runtime.h>

#define OUT_FEATURES 4096
#define IN_FEATURES  11008
#define VEC_DIM      8
#define M_TOKENS     4096   // 2 * 2048

typedef _Float16 v4h  __attribute__((ext_vector_type(4)));
typedef _Float16 v8h  __attribute__((ext_vector_type(8)));
typedef _Float16 v16h __attribute__((ext_vector_type(16)));
typedef float    v4f  __attribute__((ext_vector_type(4)));
typedef float    v8f  __attribute__((ext_vector_type(8)));

// ---------------------------------------------------------------------------
// Pass 1: dequantize weights to f16.
// One thread per 8-element codebook vector: gather codebook[idx], scale by
// per-output-row scale, write 16 bytes (v8h).  n_vectors = 4096*11008/8.
// ---------------------------------------------------------------------------
__global__ __launch_bounds__(256) void dequant_w_kernel(
    const int*   __restrict__ idx,
    const float* __restrict__ cb,
    const float* __restrict__ scales,
    _Float16*    __restrict__ w)
{
    const int VPR = IN_FEATURES / VEC_DIM;            // 1376 vectors per out-row
    size_t v = (size_t)blockIdx.x * 256 + threadIdx.x;
    int o = (int)(v / VPR);
    float s = scales[o];
    int code = idx[v];
    const v4f* c = (const v4f*)(cb + (size_t)code * VEC_DIM);
    v4f c0 = c[0];
    v4f c1 = c[1];
    v8h out;
    out[0] = (_Float16)(c0[0] * s);
    out[1] = (_Float16)(c0[1] * s);
    out[2] = (_Float16)(c0[2] * s);
    out[3] = (_Float16)(c0[3] * s);
    out[4] = (_Float16)(c1[0] * s);
    out[5] = (_Float16)(c1[1] * s);
    out[6] = (_Float16)(c1[2] * s);
    out[7] = (_Float16)(c1[3] * s);
    *(v8h*)(w + v * VEC_DIM) = out;
}

// ---------------------------------------------------------------------------
// Pass 2: convert activations f32 -> f16 (4 elements / thread, b128 in, b64 out)
// ---------------------------------------------------------------------------
__global__ __launch_bounds__(256) void cvt_x_kernel(
    const float* __restrict__ x, _Float16* __restrict__ xh)
{
    size_t i = ((size_t)blockIdx.x * 256 + threadIdx.x) * 4;
    v4f f = *(const v4f*)(x + i);
    v4h h;
    h[0] = (_Float16)f[0];
    h[1] = (_Float16)f[1];
    h[2] = (_Float16)f[2];
    h[3] = (_Float16)f[3];
    *(v4h*)(xh + i) = h;
}

// ---------------------------------------------------------------------------
// Pass 3: C[M,N] = X[M,K] * W[N,K]^T   (f16 inputs, f32 accumulate)
// Block = 256 threads = 8 waves; block tile 128x256; wave grid 2(M)x4(N);
// wave tile 64x64 = 4x4 WMMA tiles -> 16 wmma per 16 b128 loads per K-step.
// K stepped by 32 with v_wmma_f32_16x16x32_f16; K-loop unrolled x2.
// ---------------------------------------------------------------------------
__global__ __launch_bounds__(256) void gemm_f16_wmma_kernel(
    const _Float16* __restrict__ X,
    const _Float16* __restrict__ W,
    float*          __restrict__ C)
{
    const int K = IN_FEATURES;
    const int N = OUT_FEATURES;

    const int tid  = threadIdx.x;
    const int wave = tid >> 5;
    const int lane = tid & 31;
    const int lrow = lane & 15;   // row (A) / column (B) within 16x16 tile
    const int lsel = lane >> 4;   // K-half selector

    const int m0 = blockIdx.y * 128 + (wave & 1) * 64;   // 2 waves over M
    const int n0 = blockIdx.x * 256 + (wave >> 1) * 64;  // 4 waves over N

    // A-frag layout used here (contiguous k, same as B): lane holds row lrow, 16 contiguous halves at k+lsel*16
    const _Float16* abase[4];
#pragma unroll
    for (int i = 0; i < 4; ++i)
        abase[i] = X + (size_t)(m0 + i * 16 + lrow) * K + lsel * 16;

    // B-frag layout used here (contiguous k, same as A): lane holds column lrow, 16 contiguous halves at k+lsel*16
    const _Float16* bbase[4];
#pragma unroll
    for (int j = 0; j < 4; ++j)
        bbase[j] = W + (size_t)(n0 + j * 16 + lrow) * K + lsel * 16;

    v8f acc[4][4];
#pragma unroll
    for (int i = 0; i < 4; ++i)
#pragma unroll
        for (int j = 0; j < 4; ++j)
            acc[i][j] = {};

#pragma unroll 2
    for (int k = 0; k < K; k += 32) {
        v16h a[4];
        v16h b[4];
#pragma unroll
        for (int i = 0; i < 4; ++i) {
            v8h lo = *(const v8h*)(abase[i] + k);
            v8h hi = *(const v8h*)(abase[i] + k + 8);
            a[i] = __builtin_shufflevector(lo, hi, 0, 1, 2, 3, 4, 5, 6, 7,
                                                   8, 9, 10, 11, 12, 13, 14, 15);
        }
#pragma unroll
        for (int j = 0; j < 4; ++j) {
            v8h lo = *(const v8h*)(bbase[j] + k);
            v8h hi = *(const v8h*)(bbase[j] + k + 8);
            b[j] = __builtin_shufflevector(lo, hi, 0, 1, 2, 3, 4, 5, 6, 7,
                                                   8, 9, 10, 11, 12, 13, 14, 15);
        }
#pragma unroll
        for (int i = 0; i < 4; ++i)
#pragma unroll
            for (int j = 0; j < 4; ++j)
                acc[i][j] = __builtin_amdgcn_wmma_f32_16x16x32_f16(
                    /*neg_a=*/false, a[i], /*neg_b=*/false, b[j],
                    /*c_mod=*/(short)0, acc[i][j],
                    /*reuse_a=*/false, /*reuse_b=*/false);
    }

    // C/D layout: VGPR r, lane l -> (M = lsel*8 + r, N = lrow) within the tile
#pragma unroll
    for (int i = 0; i < 4; ++i) {
        const int mb = m0 + i * 16 + lsel * 8;
#pragma unroll
        for (int j = 0; j < 4; ++j) {
            const int col = n0 + j * 16 + lrow;
#pragma unroll
            for (int r = 0; r < 8; ++r)
                C[(size_t)(mb + r) * N + col] = acc[i][j][r];
        }
    }
}

// ---------------------------------------------------------------------------
extern "C" void kernel_launch(void* const* d_in, const int* in_sizes, int n_in,
                              void* d_out, int out_size, void* d_ws, size_t ws_size,
                              hipStream_t stream)
{
    const float* x        = (const float*)d_in[0];   // [2,2048,11008] f32
    const int*   indices  = (const int*)  d_in[1];   // [4096*11008/8] i32
    const float* codebook = (const float*)d_in[2];   // [32768,8] f32
    const float* scales   = (const float*)d_in[3];   // [4096,1] f32
    float*       out      = (float*)d_out;           // [2,2048,4096] f32

    // Workspace layout: Wf16 [4096*11008] then Xf16 [4096*11008] (each ~90 MB)
    _Float16* wf16 = (_Float16*)d_ws;
    _Float16* xf16 = wf16 + (size_t)OUT_FEATURES * IN_FEATURES;

    // Pass 1: dequant weights.  n_vectors = 5,636,096 = 22016 * 256
    dequant_w_kernel<<<22016, 256, 0, stream>>>(indices, codebook, scales, wf16);

    // Pass 2: convert x.  45,088,768 elems / 4 per thread / 256 = 44032 blocks
    cvt_x_kernel<<<44032, 256, 0, stream>>>(x, xf16);

    // Pass 3: GEMM.  grid = (N/256, M/128) = (16, 32)
    gemm_f16_wmma_kernel<<<dim3(16, 32), 256, 0, stream>>>(xf16, wf16, out);
}
